// CdRegressor_9904194584715
// MI455X (gfx1250) — matrix operand order repaired and checked by reading, not compiled, not run on hardware
//
#include <hip/hip_runtime.h>
#include <hip/hip_bf16.h>

typedef __attribute__((ext_vector_type(16))) _Float16 v16h;
typedef __attribute__((ext_vector_type(2)))  _Float16 h2;
typedef __attribute__((ext_vector_type(8)))  float    v8f;

union V16u { v16h v; h2 p[8]; };

#define B_  4
#define S_  80
#define P_  6500
#define F_  128

__device__ __forceinline__ float sigmoidf_(float x) {
    return 1.0f / (1.0f + __expf(-x));
}

__device__ __forceinline__ h2 relu_h2(h2 x) {
#if __has_builtin(__builtin_elementwise_max)
    return __builtin_elementwise_max(x, (h2)(_Float16)0);
#else
    h2 r;
    r.x = x.x > (_Float16)0 ? x.x : (_Float16)0;
    r.y = x.y > (_Float16)0 ? x.y : (_Float16)0;
    return r;
#endif
}

// ---------------------------------------------------------------------------
// Kernel A: per-point MLP (2->64 packed-f16 VALU, 64->128 WMMA f16) + masked
// max over P.  One block per (b,s); 4 waves, each wave owns 16 points/chunk.
// N tiles processed in 2 groups of 4 (group loop outer, P loop inner) so only
// 64 VGPRs of B fragments are live -> wave stays under 256 VGPRs.
// b2 bias is pre-loaded into the WMMA C operand (per-column constant matches
// the C layout exactly), layer-1 runs in packed f16 straight into the
// A-fragment layout: element pair (2v,2v+1) == k pair (k,k+1).
// emb[bs][f] = max_p( relu(layer2) * mask )  (== reference: relu >= 0 so the
// -inf/0 fixup collapses to a 0-initialized running max).
// ---------------------------------------------------------------------------
__global__ __launch_bounds__(128)
void point_mlp_max_kernel(const float* __restrict__ slices,
                          const float* __restrict__ pmask,
                          const float* __restrict__ W1,
                          const float* __restrict__ b1,
                          const float* __restrict__ W2,
                          const float* __restrict__ b2,
                          float* __restrict__ emb)
{
    // B-fragment-swizzled W2: [tile(8)][kfrag(2)][lane(32)][elem(16)] halves
    __shared__ __attribute__((aligned(32))) _Float16 sW2[8192];
    // layer-1 weights pre-packed as half2 in A-fragment element order:
    // index = hi*16 + kf*8 + v ;  pair covers k, k+1
    __shared__ h2 sw0[32];      // W1[:,0] pairs
    __shared__ h2 sw1[32];      // W1[:,1] pairs
    __shared__ h2 sbb[32];      // b1 pairs
    __shared__ float sb2[128];
    __shared__ float sred[4 * 128];

    const int tid  = threadIdx.x;
    const int wave = tid >> 5;
    const int lane = tid & 31;
    const int bs   = blockIdx.x;          // 0 .. B*S-1

    if (tid < 128) sb2[tid] = b2[tid];
    if (tid < 32) {
        int v   = tid & 7;
        int kf  = (tid >> 3) & 1;
        int hi2 = tid >> 4;
        int kb  = 2 * v + 16 * hi2 + 32 * kf;
        h2 w0; w0.x = (_Float16)W1[2 * kb];     w0.y = (_Float16)W1[2 * (kb + 1)];
        h2 w1; w1.x = (_Float16)W1[2 * kb + 1]; w1.y = (_Float16)W1[2 * (kb + 1) + 1];
        h2 bb; bb.x = (_Float16)b1[kb];         bb.y = (_Float16)b1[kb + 1];
        sw0[tid] = w0; sw1[tid] = w1; sbb[tid] = bb;
    }

    // Swizzle W2 (F=128 rows of 64) into B-matrix fragment layout:
    // lane L holds column n = t*16 + (L&15); k = kf*32 + e + 16*(L>=16)
    for (int idx = tid; idx < 8192; idx += 128) {
        int e  = idx & 15;
        int L  = (idx >> 4) & 31;
        int kf = (idx >> 9) & 1;
        int t  = idx >> 10;
        int k  = kf * 32 + e + ((L >= 16) ? 16 : 0);
        int n  = t * 16 + (L & 15);
        sW2[idx] = (_Float16)W2[n * 64 + k];
    }
    __syncthreads();

    const long base_bs = (long)bs * P_;
    const int  m  = lane & 15;
    const bool hi = (lane >= 16);
    const int  hb = hi ? 16 : 0;

#pragma unroll 1
    for (int g = 0; g < 2; ++g) {
        // loop-invariant B fragments for this group's 4 N-tiles (64 VGPRs)
        v16h bf[4][2];
#pragma unroll
        for (int t = 0; t < 4; ++t) {
            bf[t][0] = *(const v16h*)&sW2[(((g * 4 + t) * 2 + 0) * 32 + lane) * 16];
            bf[t][1] = *(const v16h*)&sW2[(((g * 4 + t) * 2 + 1) * 32 + lane) * 16];
        }

        float runmax[4];
#pragma unroll
        for (int t = 0; t < 4; ++t) runmax[t] = 0.0f;

        for (int p0 = wave * 16; p0 < P_; p0 += 64) {
            const int p = p0 + m;
            float x0 = 0.0f, x1 = 0.0f, pm = 0.0f;
            if (p < P_) {
                long si = (base_bs + p) * 2;
                x0 = slices[si];
                x1 = slices[si + 1];
                pm = pmask[base_bs + p];
            }
            // broadcast per-row masks for the C-tile rows this lane will see
            float maskr[8];
#pragma unroll
            for (int r = 0; r < 8; ++r)
                maskr[r] = __shfl(pm, r + (hi ? 8 : 0), 32);

            // Layer 1 (2->64) in packed f16, directly in A-fragment layout.
            const _Float16 x0h = (_Float16)x0;
            const _Float16 x1h = (_Float16)x1;
            h2 x0p; x0p.x = x0h; x0p.y = x0h;
            h2 x1p; x1p.x = x1h; x1p.y = x1h;
            V16u A0, A1;
#pragma unroll
            for (int v = 0; v < 8; ++v) {
                int i0 = hb + v;        // kf = 0  (K 0..31)
                int i1 = hb + 8 + v;    // kf = 1  (K 32..63)
                h2 t0 = x1p * sw1[i0] + (x0p * sw0[i0] + sbb[i0]);
                h2 t1 = x1p * sw1[i1] + (x0p * sw0[i1] + sbb[i1]);
                A0.p[v] = relu_h2(t0);
                A1.p[v] = relu_h2(t1);
            }

            // Layer 2 (64->128): 4 N-tiles, K=64 via 2 chained WMMAs,
            // bias folded into the C operand (per-column constant).
#pragma unroll
            for (int t = 0; t < 4; ++t) {
                const float bias = sb2[(g * 4 + t) * 16 + m];
                v8f acc = {bias, bias, bias, bias, bias, bias, bias, bias};
                acc = __builtin_amdgcn_wmma_f32_16x16x32_f16(false, A0.v, false, bf[t][0],
                                                             (short)0, acc, false, false);
                acc = __builtin_amdgcn_wmma_f32_16x16x32_f16(false, A1.v, false, bf[t][1],
                                                             (short)0, acc, false, false);
                float tmax = 0.0f;
#pragma unroll
                for (int r = 0; r < 8; ++r) {
                    float val = fmaxf(acc[r], 0.0f) * maskr[r];
                    tmax = fmaxf(tmax, val);
                }
                runmax[t] = fmaxf(runmax[t], tmax);
            }
        }

        // fold rows 0-7 (lanes 0-15) with rows 8-15 (lanes 16-31)
#pragma unroll
        for (int t = 0; t < 4; ++t)
            runmax[t] = fmaxf(runmax[t], __shfl_xor(runmax[t], 16, 32));

        if (lane < 16) {
#pragma unroll
            for (int t = 0; t < 4; ++t)
                sred[wave * 128 + (g * 4 + t) * 16 + lane] = runmax[t];
        }
    }
    __syncthreads();
    if (tid < 128) {
        float v = fmaxf(fmaxf(sred[tid], sred[128 + tid]),
                        fmaxf(sred[256 + tid], sred[384 + tid]));
        emb[bs * 128 + tid] = v;
    }
}

// ---------------------------------------------------------------------------
// Kernel B1: xg[d][s][b][g] = emb[b][s_src] . Wi_d[g] + bi_d[g] + bh_d[g]
// (s_src reversed for the backward direction).  160 blocks x 256 threads.
// ---------------------------------------------------------------------------
__global__ __launch_bounds__(256)
void lstm_gates_kernel(const float* __restrict__ emb,
                       const float* __restrict__ Wi_f, const float* __restrict__ bi_f,
                       const float* __restrict__ bh_f,
                       const float* __restrict__ Wi_b, const float* __restrict__ bi_b,
                       const float* __restrict__ bh_b,
                       float* __restrict__ xg)          // [2][S][B][512]
{
    __shared__ float semb[B_ * 128];
    const int blk   = blockIdx.x;
    const int d     = blk / S_;
    const int s     = blk % S_;
    const int s_src = d ? (S_ - 1 - s) : s;
    const float* Wi = d ? Wi_b : Wi_f;
    const float* bi = d ? bi_b : bi_f;
    const float* bh = d ? bh_b : bh_f;

    for (int i = threadIdx.x; i < B_ * 128; i += 256) {
        int b = i >> 7, f = i & 127;
        semb[i] = emb[(b * S_ + s_src) * 128 + f];
    }
    __syncthreads();

    float* out = xg + ((long)d * S_ + s) * (B_ * 512);
    for (int i = threadIdx.x; i < B_ * 512; i += 256) {
        int b = i >> 9, g = i & 511;
        const float* wr = Wi + g * 128;
        const float* er = semb + b * 128;
        float acc = 0.0f;
#pragma unroll 8
        for (int k = 0; k < 128; ++k) acc += er[k] * wr[k];
        out[b * 512 + g] = acc + bi[g] + bh[g];
    }
}

// ---------------------------------------------------------------------------
// Kernel B2: sequential LSTM scan, one block per direction, thread = gate.
// h,c,gates live in LDS; Wh streamed from L2 each step (256 KB, L2-resident).
// ---------------------------------------------------------------------------
__global__ __launch_bounds__(512)
void lstm_scan_kernel(const float* __restrict__ xg,     // [2][S][B][512]
                      const float* __restrict__ Wh_f,
                      const float* __restrict__ Wh_b,
                      float* __restrict__ hlast)        // [2][B][128]
{
    __shared__ float sh[B_ * 128];
    __shared__ float sc[B_ * 128];
    __shared__ float sg[B_ * 512];
    const int d   = blockIdx.x;
    const int tid = threadIdx.x;                        // gate index 0..511
    const float* Wh = d ? Wh_b : Wh_f;

    sh[tid] = 0.0f;
    sc[tid] = 0.0f;
    __syncthreads();

    const float* xgd  = xg + (long)d * S_ * (B_ * 512);
    const float* wrow = Wh + tid * 128;

    for (int s = 0; s < S_; ++s) {
        const float* xr = xgd + s * (B_ * 512);
        float a0 = xr[tid];
        float a1 = xr[512 + tid];
        float a2 = xr[1024 + tid];
        float a3 = xr[1536 + tid];
#pragma unroll 4
        for (int k = 0; k < 128; ++k) {
            float w = wrow[k];
            a0 += w * sh[k];
            a1 += w * sh[128 + k];
            a2 += w * sh[256 + k];
            a3 += w * sh[384 + k];
        }
        sg[tid] = a0; sg[512 + tid] = a1; sg[1024 + tid] = a2; sg[1536 + tid] = a3;
        __syncthreads();

        {   // tid -> (b,u): gates split [i | f | g | o]
            int b = tid >> 7, u = tid & 127;
            float gi = sg[b * 512 + u];
            float gf = sg[b * 512 + 128 + u];
            float gg = sg[b * 512 + 256 + u];
            float go = sg[b * 512 + 384 + u];
            float c  = sigmoidf_(gf) * sc[b * 128 + u] + sigmoidf_(gi) * tanhf(gg);
            float h  = sigmoidf_(go) * tanhf(c);
            sc[b * 128 + u] = c;
            sh[b * 128 + u] = h;
        }
        __syncthreads();
    }
    hlast[d * (B_ * 128) + tid] = sh[tid];
}

// ---------------------------------------------------------------------------
// Kernel C: head  out[b] = relu(concat(h_f,h_b) @ W3^T + b3) @ W4^T + b4
// ---------------------------------------------------------------------------
__global__ __launch_bounds__(128)
void head_kernel(const float* __restrict__ hlast,       // [2][B][128]
                 const float* __restrict__ W3, const float* __restrict__ b3,
                 const float* __restrict__ W4, const float* __restrict__ b4,
                 float* __restrict__ out)               // [B]
{
    __shared__ float sred[128];
    const int tid = threadIdx.x;                        // hidden-3 unit
    for (int b = 0; b < B_; ++b) {
        const float* hf = hlast + b * 128;
        const float* hb = hlast + B_ * 128 + b * 128;
        const float* w  = W3 + tid * 256;
        float acc = b3[tid];
#pragma unroll 4
        for (int k = 0; k < 128; ++k) acc += hf[k] * w[k];
#pragma unroll 4
        for (int k = 0; k < 128; ++k) acc += hb[k] * w[128 + k];
        sred[tid] = fmaxf(acc, 0.0f) * W4[tid];
        __syncthreads();
        for (int off = 64; off > 0; off >>= 1) {
            if (tid < off) sred[tid] += sred[tid + off];
            __syncthreads();
        }
        if (tid == 0) out[b] = sred[0] + b4[0];
        __syncthreads();
    }
}

// ---------------------------------------------------------------------------
extern "C" void kernel_launch(void* const* d_in, const int* in_sizes, int n_in,
                              void* d_out, int out_size, void* d_ws, size_t ws_size,
                              hipStream_t stream)
{
    const float* slices = (const float*)d_in[0];
    const float* pmask  = (const float*)d_in[1];
    const float* W1     = (const float*)d_in[2];
    const float* b1     = (const float*)d_in[3];
    const float* W2     = (const float*)d_in[4];
    const float* b2     = (const float*)d_in[5];
    const float* Wi_f   = (const float*)d_in[6];
    const float* Wh_f   = (const float*)d_in[7];
    const float* bi_f   = (const float*)d_in[8];
    const float* bh_f   = (const float*)d_in[9];
    const float* Wi_b   = (const float*)d_in[10];
    const float* Wh_b   = (const float*)d_in[11];
    const float* bi_b   = (const float*)d_in[12];
    const float* bh_b   = (const float*)d_in[13];
    const float* W3     = (const float*)d_in[14];
    const float* b3     = (const float*)d_in[15];
    const float* W4     = (const float*)d_in[16];
    const float* b4     = (const float*)d_in[17];
    float* out = (float*)d_out;

    // workspace layout (floats)
    float* ws    = (float*)d_ws;
    float* emb   = ws;                                  // B*S*128      = 40960
    float* xg    = emb + B_ * S_ * 128;                 // 2*S*B*512    = 327680
    float* hlast = xg + 2 * S_ * B_ * 512;              // 2*B*128      = 1024

    point_mlp_max_kernel<<<B_ * S_, 128, 0, stream>>>(slices, pmask, W1, b1, W2, b2, emb);
    lstm_gates_kernel<<<2 * S_, 256, 0, stream>>>(emb, Wi_f, bi_f, bh_f,
                                                  Wi_b, bi_b, bh_b, xg);
    lstm_scan_kernel<<<2, 512, 0, stream>>>(xg, Wh_f, Wh_b, hlast);
    head_kernel<<<1, 128, 0, stream>>>(hlast, W3, b3, W4, b4, out);
}
